// PromptVQ_82085414961198
// MI455X (gfx1250) — matrix operand order repaired and checked by reading, not compiled, not run on hardware
//
#include <hip/hip_runtime.h>
#include <hip/hip_bf16.h>
#include <stdint.h>
#include <stddef.h>

// ---------------- problem constants ----------------
#define N_ROWS   32768
#define CIN      256
#define CHID     64
#define KCODES   2048
#define NSAMP    10
#define BETA     0.9f
#define EPSV     0.01f
#define LSTRIDE  2052   // padded row stride for LDS logits (bank-conflict avoidance)

typedef __bf16  v16bf __attribute__((ext_vector_type(16)));
typedef float   v8f   __attribute__((ext_vector_type(8)));

// ---------------- workspace byte offsets ----------------
#define WS_WD    0            // 64*256  f32  (65536 B)
#define WS_WU    65536        // 256*64  f32
#define WS_CBN   131072       // 2048    f32
#define WS_PCN   139264       // 32768   f32
#define WS_CNT   270336       // 2048    f32
#define WS_EMAF  278528       // 2048    f32
#define WS_LOSS  286720       // 1       f32
#define WS_CB16  294912       // 2048*256 bf16 (1 MB), 32B aligned
#define WS_DW    1343488      // 2048*256 f32  (2 MB)
#define WS_PC    3440640      // 32768*256 f32 (32 MB)   total ~35.3 MB

// output float offsets
#define OUT_X    0
#define OUT_LOSS 8388608
#define OUT_CB   8388609
#define OUT_EMA  8912897

// deterministic counter-based uniform in (0,1)
__device__ __forceinline__ float uni01(uint32_t row, uint32_t s, uint32_t k) {
  uint64_t z = ((uint64_t)row << 32)
             ^ ((uint64_t)(s + 1) * 0x9E3779B97F4A7C15ull)
             ^ ((uint64_t)(k + 1) * 0xD1B54A32D192ED03ull)
             ^ 0x6A09E667F3BCC909ull;
  z ^= z >> 30; z *= 0xBF58476D1CE4E5B9ull;
  z ^= z >> 27; z *= 0x94D049BB133111EBull;
  z ^= z >> 31;
  uint32_t m = (uint32_t)(z >> 40);                 // 24 random bits
  return ((float)m + 0.5f) * (1.0f / 16777216.0f);  // strictly in (0,1)
}

// ---------------- kernel 0: zero the atomic accumulators ----------------
__global__ void k_init(float* __restrict__ counts, float* __restrict__ dw,
                       float* __restrict__ loss) {
  int i = blockIdx.x * blockDim.x + threadIdx.x;
  int stride = gridDim.x * blockDim.x;
  for (int k = i; k < KCODES; k += stride) counts[k] = 0.f;
  for (int k = i; k < KCODES * CIN; k += stride) dw[k] = 0.f;
  if (i == 0) loss[0] = 0.f;
}

// ---------------- kernel 1: build PHM weights (tiny) ----------------
__global__ __launch_bounds__(256)
void k_weights(const float* __restrict__ a_dn, const float* __restrict__ s_dn,
               const float* __restrict__ a_up, const float* __restrict__ s_up,
               float* __restrict__ Wd, float* __restrict__ Wu) {
  int t = threadIdx.x;
  // W_down[o][c], o<64, c<256 : o=(i,kk) i<4,kk<16 ; c=(jj,ll) jj<4,ll<64
  for (int idx = t; idx < 64 * 256; idx += 256) {
    int o = idx >> 8, c = idx & 255;
    int i = o >> 4, kk = o & 15, jj = c >> 6, ll = c & 63;
    float w = 0.f;
#pragma unroll
    for (int n = 0; n < 4; ++n)
      w += a_dn[n * 16 + i * 4 + jj] * s_dn[n * 1024 + kk * 64 + ll];
    Wd[idx] = w;
  }
  // W_up[o][c], o<256, c<64 : o=(i,kk) i<4,kk<64 ; c=(jj,ll) jj<4,ll<16
  for (int idx = t; idx < 256 * 64; idx += 256) {
    int o = idx >> 6, c = idx & 63;
    int i = o >> 6, kk = o & 63, jj = c >> 4, ll = c & 15;
    float w = 0.f;
#pragma unroll
    for (int n = 0; n < 4; ++n)
      w += a_up[n * 16 + i * 4 + jj] * s_up[n * 1024 + kk * 16 + ll];
    Wu[idx] = w;
  }
}

// ---------------- kernel 2: codebook -> bf16 + row norms ----------------
__global__ __launch_bounds__(256)
void k_codebook(const float* __restrict__ cb, __bf16* __restrict__ cb16,
                float* __restrict__ cbn) {
  __shared__ float red[8];
  int k = blockIdx.x, t = threadIdx.x;
  float v = cb[(size_t)k * CIN + t];
  cb16[(size_t)k * CIN + t] = (__bf16)v;
  float sq = v * v;
#pragma unroll
  for (int off = 16; off; off >>= 1) sq += __shfl_xor(sq, off, 32);
  if ((t & 31) == 0) red[t >> 5] = sq;
  __syncthreads();
  if (t == 0) {
    float s = 0.f;
#pragma unroll
    for (int i = 0; i < 8; ++i) s += red[i];
    cbn[k] = s;
  }
}

// ---------------- kernel 3: PHM down->relu->up, p_c + ||p_c||^2 ----------------
__global__ __launch_bounds__(256)
void k_phm(const float* __restrict__ ge, const float* __restrict__ Wd,
           const float* __restrict__ bdn, const float* __restrict__ Wu,
           const float* __restrict__ bup, float* __restrict__ p_c,
           float* __restrict__ pc_norm) {
  __shared__ float ges[256];
  __shared__ float hs[64];
  __shared__ float red[8];
  int row = blockIdx.x, t = threadIdx.x;
  ges[t] = ge[(size_t)row * CIN + t];
  __syncthreads();
  if (t < CHID) {
    float a = bdn[t];
    const float* w = Wd + t * CIN;
    for (int c = 0; c < CIN; ++c) a += w[c] * ges[c];
    hs[t] = a > 0.f ? a : 0.f;
  }
  __syncthreads();
  float a = bup[t];
  const float* w = Wu + t * CHID;
#pragma unroll
  for (int j = 0; j < CHID; ++j) a += w[j] * hs[j];
  p_c[(size_t)row * CIN + t] = a;
  float sq = a * a;
#pragma unroll
  for (int off = 16; off; off >>= 1) sq += __shfl_xor(sq, off, 32);
  if ((t & 31) == 0) red[t >> 5] = sq;
  __syncthreads();
  if (t == 0) {
    float s = 0.f;
#pragma unroll
    for (int i = 0; i < 8; ++i) s += red[i];
    pc_norm[row] = s;
  }
}

// ---------------- kernel 4: fused distances (WMMA bf16) + Gumbel-max sampling ----
// One block = 16 rows. 8 waves; wave w computes codes [256w, 256w+256) via
// 16 WMMA 16x16x32 bf16 tiles (8 K-steps each), logits held in LDS.
// Then wave w samples rows 2w, 2w+1 (10 categorical draws each, Gumbel-max).
__global__ __launch_bounds__(256)
void k_main(const float* __restrict__ x, const float* __restrict__ p_emb,
            const float* __restrict__ codebook, const float* __restrict__ p_c,
            const float* __restrict__ pc_norm, const __bf16* __restrict__ cb16,
            const float* __restrict__ cb_norm, float* __restrict__ out1,
            float* __restrict__ counts, float* __restrict__ dw,
            float* __restrict__ loss_acc) {
  extern __shared__ float smem[];
  float* logits = smem;                    // 16 * LSTRIDE
  float* pcn_s = smem + 16 * LSTRIDE;      // 16

  const int tid = threadIdx.x;
  const int lane = tid & 31;
  const int wave = tid >> 5;
  const int r0 = blockIdx.x * 16;

  if (tid < 16) pcn_s[tid] = pc_norm[r0 + tid];
  __syncthreads();

  // ---- WMMA phase: dot(p_c_tile[16x256], codebook[n][256]) ----
  const int half = lane >> 4;    // K-half selector (ISA A/B layout)
  const int mrow = lane & 15;    // A row for this lane

  // Hoist all 8 A fragments (K = 256) into registers; convert f32 -> bf16.
  v16bf aF[8];
  {
    const float* ap = p_c + (size_t)(r0 + mrow) * CIN;
#pragma unroll
    for (int kb = 0; kb < 8; ++kb) {
      int b1 = kb * 32 + half * 16;  // first contiguous 16-run
#pragma unroll
      for (int j = 0; j < 8; ++j) {
        aF[kb][j]     = (__bf16)ap[b1 + j];        // VGPR0..3 region
        aF[kb][8 + j] = (__bf16)ap[b1 + 8 + j];    // VGPR4..7 region
      }
    }
  }

  for (int t = 0; t < 16; ++t) {
    const int n0 = wave * 256 + t * 16;
    const int ncol = n0 + (lane & 15);       // B column for this lane
    const float cbn_v = cb_norm[ncol];
    // B fragment: 16 contiguous bf16 = one 32B load per K-step
    const __bf16* bp = cb16 + (size_t)ncol * CIN + half * 16;
    v8f acc = {};
#pragma unroll
    for (int kb = 0; kb < 8; ++kb) {
      v16bf bF = *(const v16bf*)(bp + kb * 32);
      acc = __builtin_amdgcn_wmma_f32_16x16x32_bf16(
          false, aF[kb], false, bF, (short)0, acc, false, false);
    }
    // C layout: VGPR r -> M = r + 8*(lane>=16), N = n0 + (lane&15)
#pragma unroll
    for (int r = 0; r < 8; ++r) {
      int m = r + 8 * half;
      float dist = pcn_s[m] + cbn_v - 2.0f * acc[r];
      logits[m * LSTRIDE + ncol] = -dist - 1e-5f;   // /TEMP with TEMP=1
    }
  }
  __syncthreads();

  // ---- sampling phase: wave w owns rows 2w, 2w+1 ----
  for (int rr = 0; rr < 2; ++rr) {
    const int m = wave * 2 + rr;
    const int row = r0 + m;
    const float* pcp = p_c + (size_t)row * CIN;
    float pcv[8], acc[8];
#pragma unroll
    for (int j = 0; j < 8; ++j) {
      pcv[j] = pcp[lane + 32 * j];
      acc[j] = 0.f;
    }
    const float* lrow = logits + m * LSTRIDE;
    for (int s = 0; s < NSAMP; ++s) {
      // Gumbel-max == categorical(logits); no softmax normalization needed
      float best = -3.4e38f;
      int bestk = 0;
      for (int t2 = 0; t2 < 64; ++t2) {
        int k = lane + 32 * t2;
        float u = uni01((uint32_t)row, (uint32_t)s, (uint32_t)k);
        float g = -__logf(-__logf(u));
        float v = lrow[k] + g;
        if (v > best || (v == best && k < bestk)) { best = v; bestk = k; }
      }
#pragma unroll
      for (int off = 16; off > 0; off >>= 1) {
        float ov = __shfl_xor(best, off, 32);
        int ok = __shfl_xor(bestk, off, 32);
        if (ov > best || (ov == best && ok < bestk)) { best = ov; bestk = ok; }
      }
      // side effects of this draw
      const float* crow = codebook + (size_t)bestk * CIN;
#pragma unroll
      for (int j = 0; j < 8; ++j) {
        acc[j] += crow[lane + 32 * j];                         // p_q accumulation
        atomicAdd(&dw[(size_t)bestk * CIN + lane + 32 * j], pcv[j]);  // samples.T @ p_c
      }
      if (lane == 0) atomicAdd(&counts[bestk], 1.0f);
    }
    // finalize row: out1 = x + p_q + p_emb ; loss accum
    float lsum = 0.f;
#pragma unroll
    for (int j = 0; j < 8; ++j) {
      int c = lane + 32 * j;
      float pq = acc[j] * (1.0f / NSAMP);
      out1[(size_t)row * CIN + c] = x[(size_t)row * CIN + c] + pq + p_emb[c];
      float d = pq - pcv[j];
      lsum += d * d;
    }
#pragma unroll
    for (int off = 16; off > 0; off >>= 1) lsum += __shfl_xor(lsum, off, 32);
    if (lane == 0) atomicAdd(loss_acc, lsum);
  }
}

// ---------------- kernel 5: EMA cluster sizes + loss scalar ----------------
__global__ __launch_bounds__(256)
void k_ema(const float* __restrict__ ema_in, const float* __restrict__ counts,
           const float* __restrict__ loss_acc, float* __restrict__ ema_fin_ws,
           float* __restrict__ out_ema, float* __restrict__ out_loss) {
  __shared__ float red[8];
  __shared__ float ntot_s;
  int t = threadIdx.x;
  float raw[8];
  float psum = 0.f;
#pragma unroll
  for (int i = 0; i < 8; ++i) {
    int k = t + i * 256;
    float r = ema_in[k] * BETA + (1.f - BETA) * (counts[k] * (1.0f / NSAMP));
    raw[i] = r;
    psum += r;
  }
#pragma unroll
  for (int off = 16; off; off >>= 1) psum += __shfl_xor(psum, off, 32);
  if ((t & 31) == 0) red[t >> 5] = psum;
  __syncthreads();
  if (t == 0) {
    float s = 0.f;
#pragma unroll
    for (int i = 0; i < 8; ++i) s += red[i];
    ntot_s = s;
  }
  __syncthreads();
  float ntot = ntot_s;
  float scale = ntot / (ntot + (float)KCODES * EPSV);
#pragma unroll
  for (int i = 0; i < 8; ++i) {
    int k = t + i * 256;
    float e = (raw[i] + EPSV) * scale;
    ema_fin_ws[k] = e;
    out_ema[k] = e;
  }
  if (t == 0) out_loss[0] = loss_acc[0] * (1.0f / (float)((size_t)N_ROWS * CIN));
}

// ---------------- kernel 6: codebook EMA update ----------------
__global__ __launch_bounds__(256)
void k_newcb(const float* __restrict__ cb, const float* __restrict__ dw,
             const float* __restrict__ ema_fin, float* __restrict__ out_cb) {
  int k = blockIdx.x, t = threadIdx.x;
  float e = ema_fin[k];
  size_t idx = (size_t)k * CIN + t;
  float d = dw[idx] * (1.0f / NSAMP);
  out_cb[idx] = cb[idx] * BETA + (1.f - BETA) * (d / e);
}

// ---------------- host launcher ----------------
extern "C" void kernel_launch(void* const* d_in, const int* in_sizes, int n_in,
                              void* d_out, int out_size, void* d_ws, size_t ws_size,
                              hipStream_t stream) {
  (void)in_sizes; (void)n_in; (void)out_size; (void)ws_size;
  const float* x        = (const float*)d_in[0];
  const float* ge       = (const float*)d_in[1];
  const float* p_emb    = (const float*)d_in[2];
  const float* codebook = (const float*)d_in[3];
  const float* ema_in   = (const float*)d_in[4];
  const float* a_dn     = (const float*)d_in[5];
  const float* s_dn     = (const float*)d_in[6];
  const float* b_dn     = (const float*)d_in[7];
  const float* a_up     = (const float*)d_in[8];
  const float* s_up     = (const float*)d_in[9];
  const float* b_up     = (const float*)d_in[10];

  char* ws = (char*)d_ws;
  float*  Wd      = (float*)(ws + WS_WD);
  float*  Wu      = (float*)(ws + WS_WU);
  float*  cbn     = (float*)(ws + WS_CBN);
  float*  pcn     = (float*)(ws + WS_PCN);
  float*  counts  = (float*)(ws + WS_CNT);
  float*  ema_fin = (float*)(ws + WS_EMAF);
  float*  loss    = (float*)(ws + WS_LOSS);
  __bf16* cb16    = (__bf16*)(ws + WS_CB16);
  float*  dw      = (float*)(ws + WS_DW);
  float*  p_c     = (float*)(ws + WS_PC);

  float* out      = (float*)d_out;
  float* out1     = out + OUT_X;
  float* out_loss = out + OUT_LOSS;
  float* out_cb   = out + OUT_CB;
  float* out_ema  = out + OUT_EMA;

  k_init<<<512, 256, 0, stream>>>(counts, dw, loss);
  k_weights<<<1, 256, 0, stream>>>(a_dn, s_dn, a_up, s_up, Wd, Wu);
  k_codebook<<<KCODES, 256, 0, stream>>>(codebook, cb16, cbn);
  k_phm<<<N_ROWS, 256, 0, stream>>>(ge, Wd, b_dn, Wu, b_up, p_c, pcn);

  size_t lds_bytes = (size_t)16 * LSTRIDE * sizeof(float) + 16 * sizeof(float);
  k_main<<<N_ROWS / 16, 256, lds_bytes, stream>>>(
      x, p_emb, codebook, p_c, pcn, cb16, cbn, out1, counts, dw, loss);

  k_ema<<<1, 256, 0, stream>>>(ema_in, counts, loss, ema_fin, out_ema, out_loss);
  k_newcb<<<KCODES, 256, 0, stream>>>(codebook, dw, ema_fin, out_cb);
}
